// GridFeatureToPointGraphConv_49435073577161
// MI455X (gfx1250) — matrix operand order repaired and checked by reading, not compiled, not run on hardware
//
#include <hip/hip_runtime.h>
#include <math.h>

// ---------------- problem constants ----------------
#define RES    48
#define NPTS   65536
#define KNN    16
#define CGRID  32
#define CPT    32
#define COUT   32
#define HID    64
#define EK     96        // e_in = 67 padded to 3*32 for f16 WMMA K-steps
#define SE_STRIDE 104    // sE row stride (halves): 52 dwords -> conflict-free half-wave b128
#define SH_STRIDE 72     // sH row stride (halves): 36 dwords -> conflict-free half-wave b128

#define WSZ_W1 (HID * EK)      // 6144 halves
#define WSZ_W2 (COUT * HID)    // 2048 halves
#define WSZ_W3 (COUT * COUT)   // 1024 halves

typedef __attribute__((ext_vector_type(16))) _Float16 v16h;
typedef __attribute__((ext_vector_type(8)))  _Float16 v8h;
typedef __attribute__((ext_vector_type(4)))  _Float16 v4h;
typedef __attribute__((ext_vector_type(8)))  float    v8f;

union V16 { v16h v; v8h h[2]; };

// Branch-free tanh-GELU. gfx1250 has native V_TANH_F32; otherwise use the
// identical sigmoid form: 0.5*(1+tanh(y)) == 1/(1+exp(-2y)).
__device__ __forceinline__ float gelu_tanh(float x) {
    const float k0 = 0.7978845608028654f;   // sqrt(2/pi)
    const float k1 = 0.044715f;
    float y = k0 * (x + k1 * x * x * x);
#if __has_builtin(__builtin_amdgcn_tanhf)
    return 0.5f * x * (1.0f + __builtin_amdgcn_tanhf(y));
#elif __has_builtin(__builtin_amdgcn_tanh_f32)
    return 0.5f * x * (1.0f + __builtin_amdgcn_tanh_f32(y));
#else
    float e = __expf(-2.0f * y);
    return x * __builtin_amdgcn_rcpf(1.0f + e);
#endif
}

// scaled grid vertex coordinate along one axis: linspace(-1,1,48)[i] * 24
__device__ __forceinline__ float gvert(int i) {
    return -24.0f + (float)i * (48.0f / 47.0f);
}

// ---------------------------------------------------------------------------
// One-time weight prep: f32 row-major -> f16 transposed [n][k], K-padded.
// Layout in ws (halves): [0,6144) W1t, [6144,8192) W2t, [8192,9216) W3t.
// ---------------------------------------------------------------------------
__global__ __launch_bounds__(256) void prep_weights_kernel(
    const float* __restrict__ W1, const float* __restrict__ W2,
    const float* __restrict__ W3, _Float16* __restrict__ wout)
{
    int t = threadIdx.x;
    for (int e = t; e < WSZ_W1; e += 256) {
        int n = e / EK, k = e - n * EK;
        wout[e] = (k < 67) ? (_Float16)W1[k * HID + n] : (_Float16)0.0f;
    }
    for (int e = t; e < WSZ_W2; e += 256) {
        int n = e / HID, k = e - n * HID;
        wout[WSZ_W1 + e] = (_Float16)W2[k * COUT + n];
    }
    for (int e = t; e < WSZ_W3; e += 256) {
        int n = e >> 5, k = e & 31;
        wout[WSZ_W1 + WSZ_W2 + e] = (_Float16)W3[k * COUT + n];
    }
}

// ---------------------------------------------------------------------------
// KNN: regular 48^3 lattice -> 16 NN are inside the surrounding 4x4x4 window.
// ---------------------------------------------------------------------------
__global__ __launch_bounds__(256) void knn16_kernel(
    const float* __restrict__ pvtx, int* __restrict__ idx_out)
{
    int m = blockIdx.x * blockDim.x + threadIdx.x;
    if (m >= NPTS) return;

    float qx = pvtx[3 * m + 0] * 24.0f;
    float qy = pvtx[3 * m + 1] * 24.0f;
    float qz = pvtx[3 * m + 2] * 24.0f;

    const float inv = 47.0f / 48.0f;
    int cx = (int)floorf((qx + 24.0f) * inv) - 1;
    int cy = (int)floorf((qy + 24.0f) * inv) - 1;
    int cz = (int)floorf((qz + 24.0f) * inv) - 1;
    cx = cx < 0 ? 0 : (cx > RES - 4 ? RES - 4 : cx);
    cy = cy < 0 ? 0 : (cy > RES - 4 ? RES - 4 : cy);
    cz = cz < 0 ? 0 : (cz > RES - 4 ? RES - 4 : cz);

    float bd[KNN];
    int   bi[KNN];
#pragma unroll
    for (int i = 0; i < KNN; ++i) { bd[i] = 3.0e38f; bi[i] = 0; }

    for (int a = 0; a < 4; ++a) {
        int ix = cx + a;
        float dx = gvert(ix) - qx; float dx2 = dx * dx;
        for (int b = 0; b < 4; ++b) {
            int iy = cy + b;
            float dy = gvert(iy) - qy; float dxy = dx2 + dy * dy;
            for (int c = 0; c < 4; ++c) {
                int iz = cz + c;
                float dz = gvert(iz) - qz;
                float d  = dxy + dz * dz;
                int   ci = (ix * RES + iy) * RES + iz;
                if (d < bd[KNN - 1]) {
                    bd[KNN - 1] = d; bi[KNN - 1] = ci;
#pragma unroll
                    for (int s = KNN - 1; s > 0; --s) {
                        if (bd[s] < bd[s - 1]) {
                            float td = bd[s]; bd[s] = bd[s - 1]; bd[s - 1] = td;
                            int   ti = bi[s]; bi[s] = bi[s - 1]; bi[s - 1] = ti;
                        }
                    }
                }
            }
        }
    }
    int4* o4 = (int4*)(idx_out + (size_t)m * KNN);
#pragma unroll
    for (int j = 0; j < 4; ++j) {
        int4 v; v.x = bi[4*j]; v.y = bi[4*j+1]; v.z = bi[4*j+2]; v.w = bi[4*j+3];
        o4[j] = v;
    }
}

// ---------------------------------------------------------------------------
// Fused edge-MLP: 128 threads (4 waves) handle 8 points = 128 edges.
// ---------------------------------------------------------------------------
__global__ __launch_bounds__(128) void graphconv_kernel(
    const float* __restrict__ gvtx,  const float* __restrict__ gfeat,
    const float* __restrict__ pvtx,  const float* __restrict__ pfeat,
    const _Float16* __restrict__ wts,
    const float* __restrict__ B1, const float* __restrict__ B2,
    const float* __restrict__ B3,
    const int*   __restrict__ nn,    float* __restrict__ out)
{
    __shared__ __align__(16) _Float16 sE[128 * SE_STRIDE];   // edge features f16
    __shared__ __align__(16) _Float16 sH[128 * SH_STRIDE];   // layer1 out f16
    __shared__ __align__(16) _Float16 sW1[WSZ_W1];           // W1^T [n][k]
    __shared__ __align__(16) _Float16 sW2[WSZ_W2];           // W2^T [n][k]
    __shared__ __align__(16) _Float16 sW3[WSZ_W3];           // W3^T [n][k]
    __shared__ __align__(16) _Float16 sG[16 * COUT];         // mean features f16
    __shared__ float sB1[HID], sB2[COUT], sB3[COUT];

    const int tid  = threadIdx.x;
    const int lane = tid & 31;
    const int wv   = tid >> 5;        // wave 0..3
    const int nlo  = lane & 15;
    const int hsel = lane >> 4;       // lane half select

    // ---- pre-transposed f16 weights -> LDS (pure vector copies) ----
#pragma unroll
    for (int e = tid * 8; e < WSZ_W1; e += 128 * 8)
        *(v8h*)&sW1[e] = *(const v8h*)(wts + e);
#pragma unroll
    for (int e = tid * 8; e < WSZ_W2; e += 128 * 8)
        *(v8h*)&sW2[e] = *(const v8h*)(wts + WSZ_W1 + e);
    if (tid * 8 < WSZ_W3)
        *(v8h*)&sW3[tid * 8] = *(const v8h*)(wts + WSZ_W1 + WSZ_W2 + tid * 8);
    if (tid < HID)  sB1[tid] = B1[tid];
    if (tid < COUT) { sB2[tid] = B2[tid]; sB3[tid] = B3[tid]; }
    if (tid < 16 * COUT / 8) {
        v8h z = {};
        *(v8h*)&sG[tid * 8] = z;
    }

    // ---- build edge features: edge tid = (local point p = tid>>4, nbr tid&15)
    {
        int p  = tid >> 4;
        int mg = blockIdx.x * 8 + p;
        int gi = nn[mg * KNN + (tid & 15)];
        float qx = pvtx[3 * mg + 0] * 24.0f;
        float qy = pvtx[3 * mg + 1] * 24.0f;
        float qz = pvtx[3 * mg + 2] * 24.0f;

        _Float16* er = &sE[tid * SE_STRIDE];
        const float4* g4 = (const float4*)(gfeat + (size_t)gi * CGRID);
#pragma unroll
        for (int j = 0; j < 8; ++j) {
            float4 v = g4[j];
            v4h h; h.x = (_Float16)v.x; h.y = (_Float16)v.y;
            h.z = (_Float16)v.z; h.w = (_Float16)v.w;
            *(v4h*)(er + 4 * j) = h;
        }
        const float4* p4 = (const float4*)(pfeat + (size_t)mg * CPT);
#pragma unroll
        for (int j = 0; j < 8; ++j) {
            float4 v = p4[j];
            v4h h; h.x = (_Float16)v.x; h.y = (_Float16)v.y;
            h.z = (_Float16)v.z; h.w = (_Float16)v.w;
            *(v4h*)(er + 32 + 4 * j) = h;
        }
        v4h r; r.x = (_Float16)(gvtx[3 * gi + 0] * 24.0f - qx);
        r.y = (_Float16)(gvtx[3 * gi + 1] * 24.0f - qy);
        r.z = (_Float16)(gvtx[3 * gi + 2] * 24.0f - qz);
        r.w = (_Float16)0.0f;
        *(v4h*)(er + 64) = r;
        v4h z4 = {};
        *(v4h*)(er + 68) = z4;
        v8h z8 = {};
#pragma unroll
        for (int j = 72; j < EK; j += 8) *(v8h*)(er + j) = z8;
    }
    __syncthreads();

    // ---- per-wave: two 16-edge tiles; each tile == one point's 16 neighbors
    for (int tt = wv; tt < 8; tt += 4) {
        // ===== layer 1: (16 x 96) @ (96 x 64) via 3 K-steps x 4 N-tiles =====
        const _Float16* Arow = &sE[(tt * 16 + nlo) * SE_STRIDE];
        v8f acc[4] = {};
#pragma unroll
        for (int kb = 0; kb < EK; kb += 32) {
            V16 a;
            a.h[0] = *(const v8h*)(Arow + kb + hsel * 16);
            a.h[1] = *(const v8h*)(Arow + kb + hsel * 16 + 8);
#pragma unroll
            for (int nt = 0; nt < 4; ++nt) {
                V16 bm;
                const _Float16* Brow = &sW1[(nt * 16 + nlo) * EK + kb + hsel * 16];
                bm.h[0] = *(const v8h*)(Brow);
                bm.h[1] = *(const v8h*)(Brow + 8);
                acc[nt] = __builtin_amdgcn_wmma_f32_16x16x32_f16(
                    false, a.v, false, bm.v, (short)0, acc[nt], false, false);
            }
        }
        // bias + GELU -> sH f16
#pragma unroll
        for (int nt = 0; nt < 4; ++nt) {
            int col = nt * 16 + nlo;
            float bb = sB1[col];
#pragma unroll
            for (int rr = 0; rr < 8; ++rr) {
                int M = rr + 8 * hsel;
                sH[(tt * 16 + M) * SH_STRIDE + col] =
                    (_Float16)gelu_tanh(acc[nt][rr] + bb);
            }
        }

        // ===== layer 2: (16 x 64) @ (64 x 32) via 2 K-steps x 2 N-tiles =====
        const _Float16* A2row = &sH[(tt * 16 + nlo) * SH_STRIDE];
        v8f acc2[2] = {};
#pragma unroll
        for (int kb = 0; kb < HID; kb += 32) {
            V16 a2;
            a2.h[0] = *(const v8h*)(A2row + kb + hsel * 16);
            a2.h[1] = *(const v8h*)(A2row + kb + hsel * 16 + 8);
#pragma unroll
            for (int nt = 0; nt < 2; ++nt) {
                V16 bm;
                const _Float16* Brow = &sW2[(nt * 16 + nlo) * HID + kb + hsel * 16];
                bm.h[0] = *(const v8h*)(Brow);
                bm.h[1] = *(const v8h*)(Brow + 8);
                acc2[nt] = __builtin_amdgcn_wmma_f32_16x16x32_f16(
                    false, a2.v, false, bm.v, (short)0, acc2[nt], false, false);
            }
        }
        // bias + GELU + mean over the tile's 16 rows (this point's neighbors)
#pragma unroll
        for (int nt = 0; nt < 2; ++nt) {
            int col = nt * 16 + nlo;
            float bb = sB2[col];
            float s = 0.0f;
#pragma unroll
            for (int rr = 0; rr < 8; ++rr) s += gelu_tanh(acc2[nt][rr] + bb);
            s += __shfl_xor(s, 16, 32);          // combine row halves
            if (hsel == 0) sG[tt * COUT + col] = (_Float16)(s * 0.0625f);
        }
    }
    __syncthreads();

    // ===== layer 3: (16 x 32) @ (32 x 32), wave 0, rows 8..15 are zero =====
    if (wv == 0) {
        V16 a3;
        a3.h[0] = *(const v8h*)&sG[nlo * COUT + hsel * 16];
        a3.h[1] = *(const v8h*)&sG[nlo * COUT + hsel * 16 + 8];
#pragma unroll
        for (int nt = 0; nt < 2; ++nt) {
            V16 bm;
            const _Float16* Brow = &sW3[(nt * 16 + nlo) * COUT + hsel * 16];
            bm.h[0] = *(const v8h*)(Brow);
            bm.h[1] = *(const v8h*)(Brow + 8);
            v8f z = {};
            v8f d = __builtin_amdgcn_wmma_f32_16x16x32_f16(
                false, a3.v, false, bm.v, (short)0, z, false, false);
            if (hsel == 0) {                      // keep rows M < 8
                int col = nt * 16 + nlo;
                float bb = sB3[col];
#pragma unroll
                for (int rr = 0; rr < 8; ++rr) {
                    out[((size_t)blockIdx.x * 8 + rr) * COUT + col] =
                        gelu_tanh(d[rr] + bb);
                }
            }
        }
    }
}

extern "C" void kernel_launch(void* const* d_in, const int* in_sizes, int n_in,
                              void* d_out, int out_size, void* d_ws, size_t ws_size,
                              hipStream_t stream)
{
    const float* gvtx  = (const float*)d_in[0];
    const float* gfeat = (const float*)d_in[1];
    const float* pvtx  = (const float*)d_in[2];
    const float* pfeat = (const float*)d_in[3];
    const float* W1    = (const float*)d_in[4];
    const float* b1    = (const float*)d_in[5];
    const float* W2    = (const float*)d_in[6];
    const float* b2    = (const float*)d_in[7];
    const float* W3    = (const float*)d_in[8];
    const float* b3    = (const float*)d_in[9];
    float* out = (float*)d_out;

    int*      idx = (int*)d_ws;                        // 4 MB: 65536*16 int32
    _Float16* wts = (_Float16*)((char*)d_ws + (size_t)NPTS * KNN * sizeof(int));

    prep_weights_kernel<<<1, 256, 0, stream>>>(W1, W2, W3, wts);
    knn16_kernel<<<NPTS / 256, 256, 0, stream>>>(pvtx, idx);
    graphconv_kernel<<<NPTS / 8, 128, 0, stream>>>(
        gvtx, gfeat, pvtx, pfeat, wts, b1, b2, b3, idx, out);
}
